// EfficientSelfAttention_57028575756691
// MI455X (gfx1250) — matrix operand order repaired and checked by reading, not compiled, not run on hardware
//
#include <hip/hip_runtime.h>
#include <hip/hip_bf16.h>

typedef __attribute__((ext_vector_type(16))) __bf16 v16bf;
typedef __attribute__((ext_vector_type(8)))  __bf16 v8bf;
typedef __attribute__((ext_vector_type(4)))  __bf16 v4bf;
typedef __attribute__((ext_vector_type(8)))  float  v8f;
typedef __attribute__((ext_vector_type(4)))  unsigned v4u;
typedef __attribute__((ext_vector_type(8)))  int    v8i;
typedef __attribute__((ext_vector_type(4)))  int    v4i;

#define BT   32768   // B*T
#define DIM  1024
#define NH   16
#define HD   64

// ---- workspace layout (bytes) ----
static const size_t OFF_XN = 0;                        // 32768*1024 bf16 = 64MB
static const size_t OFF_WQ = 67108864;                 // 1024*1024 bf16 = 2MB
static const size_t OFF_WK = 69206016;
static const size_t OFF_WV = 71303168;
static const size_t OFF_Q  = 73400320;                 // softmaxed q, bf16, 64MB
static const size_t OFF_K  = 140509184;                // raw k, bf16, 64MB
static const size_t OFF_V  = 207618048;                // raw v, bf16, 64MB
static const size_t OFF_AT = 274726912;                // attn^T per (b,h): 512*64*64 bf16 = 4MB

// -------- WMMA fragment loaders (row-major [rows][ldk] source) --------
__device__ __forceinline__ v16bf frag_a(const __bf16* base, int ldk, int lane) {
  const __bf16* p = base + (lane & 15) * ldk + ((lane >> 4) << 3);
  v8bf lo = *(const v8bf*)p;
  v8bf hi = *(const v8bf*)(p + 16);
  return __builtin_shufflevector(lo, hi, 0,1,2,3,4,5,6,7,8,9,10,11,12,13,14,15);
}
__device__ __forceinline__ v16bf frag_b(const __bf16* base, int ldk, int lane) {
  const __bf16* p = base + (lane & 15) * ldk + ((lane >> 4) << 3);
  v8bf lo = *(const v8bf*)p;
  v8bf hi = *(const v8bf*)(p + 16);
  return __builtin_shufflevector(lo, hi, 0,1,2,3,4,5,6,7,8,9,10,11,12,13,14,15);
}

__device__ __forceinline__ v8f wmma_bf16(v16bf a, v16bf b, v8f c) {
  return __builtin_amdgcn_wmma_f32_16x16x32_bf16(false, a, false, b, (short)0, c, false, false);
}

// -------- Tensor Data Mover: 2D tile (tile_dim0=32 bf16 elems/row) global -> LDS --------
// D# group0: count=1 | lds_addr | global_addr[56:0] | type=2
// D# group1: data_size=1(2B); tensor_dim0=1024; tensor_dim1=0x100000;
//            tile_dim0=32; tile_dim1=rows; tensor_dim0_stride=1024 elems
__device__ __forceinline__ void tdm_load_2d(unsigned long long gaddr, unsigned lds_addr,
                                            unsigned rows) {
  v4u g0;
  g0[0] = 1u;                                                    // count=1, user mode
  g0[1] = lds_addr;                                              // LDS byte offset
  g0[2] = (unsigned)gaddr;                                       // global_addr[31:0]
  g0[3] = ((unsigned)(gaddr >> 32) & 0x01FFFFFFu) | 0x80000000u; // addr[56:32] | type=2
  v8i g1;
  g1[0] = 0x00010000;                    // data_size=1 (2-byte elements)
  g1[1] = (int)(1024u << 16);            // tensor_dim0 = 1024 (lo16 -> bits[31:16])
  g1[2] = 0;                             // tensor_dim0 hi16=0, tensor_dim1 lo16=0
  g1[3] = (int)(0x10u | (32u << 16));    // tensor_dim1 hi16 (0x100000>>16), tile_dim0=32
  g1[4] = (int)rows;                     // tile_dim1 = rows, tile_dim2 = 0
  g1[5] = 1024;                          // tensor_dim0_stride[31:0] = 1024 elems
  g1[6] = 0;                             // stride hi, tensor_dim1_stride lo
  g1[7] = 0;
  v4i z = {0, 0, 0, 0};
#if __clang_major__ >= 23
  v8i z8 = {0, 0, 0, 0, 0, 0, 0, 0};
  __builtin_amdgcn_tensor_load_to_lds(g0, g1, z, z, z8, 0);
#else
  __builtin_amdgcn_tensor_load_to_lds(g0, g1, z, z, 0);
#endif
}

// -------- 1) LayerNorm f32 -> bf16 --------
__launch_bounds__(256)
__global__ void ln_kernel(const float* __restrict__ x, const float* __restrict__ g,
                          const float* __restrict__ bsh, __bf16* __restrict__ xn) {
  __shared__ float part[8];
  const int row = blockIdx.x, tid = threadIdx.x, lane = tid & 31, wv = tid >> 5;
  float4 v = ((const float4*)(x + (size_t)row * DIM))[tid];
  float s = v.x + v.y + v.z + v.w;
  #pragma unroll
  for (int o = 16; o > 0; o >>= 1) s += __shfl_down(s, o, 32);
  if (lane == 0) part[wv] = s;
  __syncthreads();
  float tot = 0.f;
  #pragma unroll
  for (int i = 0; i < 8; ++i) tot += part[i];
  const float mu = tot * (1.0f / DIM);
  __syncthreads();
  float d0 = v.x - mu, d1 = v.y - mu, d2 = v.z - mu, d3 = v.w - mu;
  float sq = d0*d0 + d1*d1 + d2*d2 + d3*d3;
  #pragma unroll
  for (int o = 16; o > 0; o >>= 1) sq += __shfl_down(sq, o, 32);
  if (lane == 0) part[wv] = sq;
  __syncthreads();
  float v2 = 0.f;
  #pragma unroll
  for (int i = 0; i < 8; ++i) v2 += part[i];
  const float rs = rsqrtf(v2 * (1.0f / DIM) + 1e-5f);
  float4 gv = ((const float4*)g)[tid];
  float4 bv = ((const float4*)bsh)[tid];
  v4bf o;
  o[0] = (__bf16)(d0 * rs * gv.x + bv.x);
  o[1] = (__bf16)(d1 * rs * gv.y + bv.y);
  o[2] = (__bf16)(d2 * rs * gv.z + bv.z);
  o[3] = (__bf16)(d3 * rs * gv.w + bv.w);
  *(v4bf*)(xn + (size_t)row * DIM + tid * 4) = o;
}

// -------- 2) f32 -> bf16 weight convert --------
__launch_bounds__(256)
__global__ void cvt_kernel(const float* __restrict__ s, __bf16* __restrict__ d, int n4) {
  int i = blockIdx.x * 256 + threadIdx.x;
  if (i < n4) {
    float4 v = ((const float4*)s)[i];
    v4bf o; o[0] = (__bf16)v.x; o[1] = (__bf16)v.y; o[2] = (__bf16)v.z; o[3] = (__bf16)v.w;
    ((v4bf*)d)[i] = o;
  }
}

// -------- 3) QKV GEMM via TDM double-buffering: dst = xn @ W^T + bias --------
// mode 2 adds the fused head-dim softmax epilogue (BLOCK_N == head dim == 64).
__launch_bounds__(256)
__global__ void qkv_gemm(const __bf16* __restrict__ xn, const __bf16* __restrict__ w,
                         const float* __restrict__ bias, __bf16* __restrict__ dst, int mode) {
  __shared__ __align__(16) char smem_raw[128 * 64 * 4];  // 32KB: 2x(A 8KB + B 4KB), epilogue 32KB
  float* Cs = (float*)smem_raw;

  const int tid = threadIdx.x, lane = tid & 31, wv = tid >> 5;
  const int row0 = blockIdx.x * 128, col0 = blockIdx.y * 64;
  const int m0 = (wv >> 1) * 32, n0 = (wv & 1) * 32;

  v8f acc[2][2] = {};

  const unsigned lds0 = (unsigned)(size_t)(void*)smem_raw;   // LDS byte offset (addr[31:0])
  const unsigned long long gaA = (unsigned long long)(size_t)(xn + (size_t)row0 * DIM);
  const unsigned long long gaB = (unsigned long long)(size_t)(w + (size_t)col0 * DIM);

  // prologue: fill both buffers (A tiles 128x32, B tiles 64x32), 2 tensor ops per buffer
  if (wv == 0) {
    tdm_load_2d(gaA, lds0, 128);
    tdm_load_2d(gaB, lds0 + 8192, 64);
    tdm_load_2d(gaA + 64, lds0 + 12288, 128);
    tdm_load_2d(gaB + 64, lds0 + 20480, 64);
  }

  for (int i = 0; i < 31; ++i) {
    const int sel = i & 1;
    if (wv == 0) __builtin_amdgcn_s_wait_tensorcnt((short)2);  // buffer `sel` complete
    __syncthreads();
    {
      const __bf16* As = (const __bf16*)(smem_raw + sel * 12288);
      const __bf16* Bs = (const __bf16*)(smem_raw + sel * 12288 + 8192);
      v16bf af0 = frag_a(As + (m0)      * 32, 32, lane);
      v16bf af1 = frag_a(As + (m0 + 16) * 32, 32, lane);
      v16bf bf0 = frag_b(Bs + (n0)      * 32, 32, lane);
      v16bf bf1 = frag_b(Bs + (n0 + 16) * 32, 32, lane);
      acc[0][0] = wmma_bf16(af0, bf0, acc[0][0]);
      acc[0][1] = wmma_bf16(af0, bf1, acc[0][1]);
      acc[1][0] = wmma_bf16(af1, bf0, acc[1][0]);
      acc[1][1] = wmma_bf16(af1, bf1, acc[1][1]);
    }
    __syncthreads();   // all waves drained buffer `sel`
    if (wv == 0 && i + 2 < 32) {
      const unsigned long long koff = (unsigned long long)(i + 2) * 64;  // bytes (32 elems)
      tdm_load_2d(gaA + koff, lds0 + sel * 12288, 128);
      tdm_load_2d(gaB + koff, lds0 + sel * 12288 + 8192, 64);
    }
  }
  // peeled last iteration (i = 31, sel = 1)
  if (wv == 0) __builtin_amdgcn_s_wait_tensorcnt((short)0);
  __syncthreads();
  {
    const __bf16* As = (const __bf16*)(smem_raw + 12288);
    const __bf16* Bs = (const __bf16*)(smem_raw + 12288 + 8192);
    v16bf af0 = frag_a(As + (m0)      * 32, 32, lane);
    v16bf af1 = frag_a(As + (m0 + 16) * 32, 32, lane);
    v16bf bf0 = frag_b(Bs + (n0)      * 32, 32, lane);
    v16bf bf1 = frag_b(Bs + (n0 + 16) * 32, 32, lane);
    acc[0][0] = wmma_bf16(af0, bf0, acc[0][0]);
    acc[0][1] = wmma_bf16(af0, bf1, acc[0][1]);
    acc[1][0] = wmma_bf16(af1, bf0, acc[1][0]);
    acc[1][1] = wmma_bf16(af1, bf1, acc[1][1]);
  }
  __syncthreads();   // protect LDS reuse by epilogue

  if (mode != 2) {
    #pragma unroll
    for (int mi = 0; mi < 2; ++mi)
      #pragma unroll
      for (int ni = 0; ni < 2; ++ni) {
        const int col = col0 + n0 + ni * 16 + (lane & 15);
        const float bb = bias[col];
        #pragma unroll
        for (int r = 0; r < 8; ++r) {
          const int row = row0 + m0 + mi * 16 + r + ((lane >> 4) << 3);
          dst[(size_t)row * DIM + col] = (__bf16)(acc[mi][ni][r] + bb);
        }
      }
  } else {
    #pragma unroll
    for (int mi = 0; mi < 2; ++mi)
      #pragma unroll
      for (int ni = 0; ni < 2; ++ni) {
        const int c = n0 + ni * 16 + (lane & 15);
        const float bb = bias[col0 + c];
        #pragma unroll
        for (int r = 0; r < 8; ++r) {
          const int rr = m0 + mi * 16 + r + ((lane >> 4) << 3);
          Cs[rr * 64 + c] = acc[mi][ni][r] + bb;
        }
      }
    __syncthreads();
    if (tid < 128) {
      const float* rowp = Cs + tid * 64;
      float mx = -1e30f;
      #pragma unroll 8
      for (int c = 0; c < 64; ++c) mx = fmaxf(mx, rowp[c]);
      float s = 0.f;
      #pragma unroll 8
      for (int c = 0; c < 64; ++c) s += __expf(rowp[c] - mx);
      const float inv = 1.0f / s;
      __bf16* op = dst + (size_t)(row0 + tid) * DIM + col0;
      #pragma unroll 8
      for (int c = 0; c < 64; ++c) op[c] = (__bf16)(__expf(rowp[c] - mx) * inv);
    }
  }
}

// -------- 4) attn^T = (softmax_T(k))^T @ v per (b,h), stored [l][d] bf16 --------
__launch_bounds__(256)
__global__ void attn_kernel(const __bf16* __restrict__ kbf, const __bf16* __restrict__ vbf,
                            __bf16* __restrict__ attnT) {
  __shared__ __bf16 eT[64 * 32];
  __shared__ __bf16 vT[64 * 32];
  __shared__ float colmax[64], Zl[64], rz[64], smax[256];

  const int b = blockIdx.x >> 4, h = blockIdx.x & 15;
  const int tid = threadIdx.x, lane = tid & 31, wv = tid >> 5;
  const size_t kbase = ((size_t)b * 1024) * DIM + h * HD;

  { // column max over T (per head dim)
    const int c = tid & 63, seg = tid >> 6;
    float m = -1e30f;
    for (int n = seg * 256; n < seg * 256 + 256; ++n)
      m = fmaxf(m, (float)kbf[kbase + (size_t)n * DIM + c]);
    smax[tid] = m;
  }
  if (tid < 64) Zl[tid] = 0.f;
  __syncthreads();
  if (tid < 64)
    colmax[tid] = fmaxf(fmaxf(smax[tid], smax[tid + 64]), fmaxf(smax[tid + 128], smax[tid + 192]));
  __syncthreads();

  const int d0 = (wv >> 1) * 16;
  const int l0 = (wv & 1) * 32;
  v8f acc[2] = {};
  const int sn = tid >> 3, sc = (tid & 7) * 8;

  for (int n0 = 0; n0 < 1024; n0 += 32) {
    v8bf kv = *(const v8bf*)(kbf + kbase + (size_t)(n0 + sn) * DIM + sc);
    v8bf vv = *(const v8bf*)(vbf + kbase + (size_t)(n0 + sn) * DIM + sc);
    #pragma unroll
    for (int j = 0; j < 8; ++j) {
      const float e = __expf((float)kv[j] - colmax[sc + j]);
      atomicAdd(&Zl[sc + j], e);          // ds_add_f32
      eT[(sc + j) * 32 + sn] = (__bf16)e; // transpose into LDS
      vT[(sc + j) * 32 + sn] = vv[j];
    }
    __syncthreads();
    v16bf af  = frag_a(eT + d0 * 32, 32, lane);
    v16bf bf0 = frag_b(vT + (l0)      * 32, 32, lane);
    v16bf bf1 = frag_b(vT + (l0 + 16) * 32, 32, lane);
    acc[0] = wmma_bf16(af, bf0, acc[0]);
    acc[1] = wmma_bf16(af, bf1, acc[1]);
    __syncthreads();
  }

  if (tid < 64) rz[tid] = 1.0f / Zl[tid];
  __syncthreads();
  __bf16* ab = attnT + (size_t)(b * NH + h) * HD * HD;
  #pragma unroll
  for (int t = 0; t < 2; ++t)
    #pragma unroll
    for (int r = 0; r < 8; ++r) {
      const int d = d0 + r + ((lane >> 4) << 3);
      const int l = l0 + t * 16 + (lane & 15);
      ab[l * HD + d] = (__bf16)(acc[t][r] * rz[d]);
    }
}

// -------- 5) out = x + q_sm @ attn  (attn^T stored [l][d] == B operand) --------
__launch_bounds__(256)
__global__ void y_out_kernel(const float* __restrict__ x, const __bf16* __restrict__ qbf,
                             const __bf16* __restrict__ attnT, float* __restrict__ out) {
  __shared__ __bf16 qs[128 * 64];
  __shared__ __bf16 as_[64 * 64];
  const int tid = threadIdx.x, lane = tid & 31, wv = tid >> 5;
  const int row0 = blockIdx.x * 128, b = row0 >> 10;
  const int m0 = (wv >> 1) * 32, n0 = (wv & 1) * 32;
  const int qrow = tid >> 1, qc = (tid & 1) * 32;
  const int arow = tid >> 2, ac = (tid & 3) * 16;

  for (int h = 0; h < NH; ++h) {
    const __bf16* qg = qbf + (size_t)(row0 + qrow) * DIM + h * HD + qc;
    const __bf16* ag = attnT + ((size_t)(b * NH + h) * HD + arow) * HD + ac;
    v8bf q0 = *(const v8bf*)(qg);      v8bf q1 = *(const v8bf*)(qg + 8);
    v8bf q2 = *(const v8bf*)(qg + 16); v8bf q3 = *(const v8bf*)(qg + 24);
    v8bf a0 = *(const v8bf*)(ag);      v8bf a1 = *(const v8bf*)(ag + 8);
    *(v8bf*)(qs + qrow * 64 + qc)      = q0;
    *(v8bf*)(qs + qrow * 64 + qc + 8)  = q1;
    *(v8bf*)(qs + qrow * 64 + qc + 16) = q2;
    *(v8bf*)(qs + qrow * 64 + qc + 24) = q3;
    *(v8bf*)(as_ + arow * 64 + ac)     = a0;
    *(v8bf*)(as_ + arow * 64 + ac + 8) = a1;
    __syncthreads();

    v8f acc[2][2] = {};
    #pragma unroll
    for (int k0 = 0; k0 < 64; k0 += 32) {
      v16bf af0 = frag_a(qs + (m0)      * 64 + k0, 64, lane);
      v16bf af1 = frag_a(qs + (m0 + 16) * 64 + k0, 64, lane);
      v16bf bf0 = frag_b(as_ + (n0)      * 64 + k0, 64, lane);
      v16bf bf1 = frag_b(as_ + (n0 + 16) * 64 + k0, 64, lane);
      acc[0][0] = wmma_bf16(af0, bf0, acc[0][0]);
      acc[0][1] = wmma_bf16(af0, bf1, acc[0][1]);
      acc[1][0] = wmma_bf16(af1, bf0, acc[1][0]);
      acc[1][1] = wmma_bf16(af1, bf1, acc[1][1]);
    }
    #pragma unroll
    for (int mi = 0; mi < 2; ++mi)
      #pragma unroll
      for (int ni = 0; ni < 2; ++ni)
        #pragma unroll
        for (int r = 0; r < 8; ++r) {
          const int row = row0 + m0 + mi * 16 + r + ((lane >> 4) << 3);
          const int col = h * HD + n0 + ni * 16 + (lane & 15);
          const size_t idx = (size_t)row * DIM + col;
          out[idx] = x[idx] + acc[mi][ni][r];
        }
    __syncthreads();
  }
}

extern "C" void kernel_launch(void* const* d_in, const int* in_sizes, int n_in,
                              void* d_out, int out_size, void* d_ws, size_t ws_size,
                              hipStream_t stream) {
  const float* x    = (const float*)d_in[0];
  const float* ln_g = (const float*)d_in[1];
  const float* ln_b = (const float*)d_in[2];
  const float* Wq   = (const float*)d_in[3];
  const float* bq   = (const float*)d_in[4];
  const float* Wk   = (const float*)d_in[5];
  const float* bk   = (const float*)d_in[6];
  const float* Wv   = (const float*)d_in[7];
  const float* bv   = (const float*)d_in[8];

  char* ws = (char*)d_ws;
  __bf16* xn  = (__bf16*)(ws + OFF_XN);
  __bf16* wqb = (__bf16*)(ws + OFF_WQ);
  __bf16* wkb = (__bf16*)(ws + OFF_WK);
  __bf16* wvb = (__bf16*)(ws + OFF_WV);
  __bf16* qbf = (__bf16*)(ws + OFF_Q);
  __bf16* kbf = (__bf16*)(ws + OFF_K);
  __bf16* vbf = (__bf16*)(ws + OFF_V);
  __bf16* at  = (__bf16*)(ws + OFF_AT);

  ln_kernel<<<BT, 256, 0, stream>>>(x, ln_g, ln_b, xn);
  cvt_kernel<<<1024, 256, 0, stream>>>(Wq, wqb, DIM * DIM / 4);
  cvt_kernel<<<1024, 256, 0, stream>>>(Wk, wkb, DIM * DIM / 4);
  cvt_kernel<<<1024, 256, 0, stream>>>(Wv, wvb, DIM * DIM / 4);

  dim3 gg(BT / 128, DIM / 64);
  qkv_gemm<<<gg, 256, 0, stream>>>(xn, wkb, bk, kbf, 0);
  qkv_gemm<<<gg, 256, 0, stream>>>(xn, wvb, bv, vbf, 1);
  qkv_gemm<<<gg, 256, 0, stream>>>(xn, wqb, bq, qbf, 2);

  attn_kernel<<<32 * NH, 256, 0, stream>>>(kbf, vbf, at);
  y_out_kernel<<<BT / 128, 256, 0, stream>>>(x, qbf, at, (float*)d_out);
}
